// MYLSTM_36971078483991
// MI455X (gfx1250) — matrix operand order repaired and checked by reading, not compiled, not run on hardware
//
#include <hip/hip_runtime.h>
#include <hip/hip_bf16.h>

// ---------------------------------------------------------------------------
// 2-layer LSTM for MI455X (gfx1250), wave32 + bf16 WMMA.
//   Phase A: i2h = X @ Wi^T + bi   (full-chip tiled WMMA GEMM, bf16->f32)
//   Phase B: sequential scan; Wh slice resident in 256KB LDS per WGP,
//            c-state in WMMA C-fragments, h double-buffered via L2 with a
//            16-block spin barrier per timestep.
// ---------------------------------------------------------------------------

typedef __bf16 bf16_t;
typedef __attribute__((ext_vector_type(16))) __bf16 v16bf;
typedef __attribute__((ext_vector_type(8)))  __bf16 v8bf;
typedef __attribute__((ext_vector_type(8)))  float  v8f;

#define B_  32
#define T_  2048
#define D_  512
#define H_  512
#define G4_ 2048  /* 4*H */

// --------------------------- WMMA helpers ----------------------------------

__device__ __forceinline__ v8f wmma_bf16(v16bf a, v16bf b, v8f c) {
  // (neg_a, A, neg_b, B, c_mod, C, reuse_a, reuse_b)
  return __builtin_amdgcn_wmma_f32_16x16x32_bf16(false, a, false, b,
                                                 (short)0, c, false, false);
}

// A-fragment (16x32 bf16, row-major source with leading dim `lda`):
// lanes 0-15: M=lane, K = k0+{0..15}; lanes 16-31: K = k0+{16..31}
__device__ __forceinline__ v16bf load_a_frag(const bf16_t* __restrict__ A,
                                             int lda, int m0, int k0, int lane) {
  int m  = m0 + (lane & 15);
  int kh = (lane >> 4) << 4;  // 0 or 16
  const bf16_t* p = A + (size_t)m * lda + (k0 + kh);
  v8bf lo = *(const v8bf*)p;
  v8bf hi = *(const v8bf*)(p + 8);
  return __builtin_shufflevector(lo, hi, 0,1,2,3,4,5,6,7,8,9,10,11,12,13,14,15);
}

// B-fragment (32x16 bf16) from W stored [N][K] row-major (i.e. W as given,
// since gates = x @ W^T). lane: n = n0+(lane&15), 16 contiguous k's.
__device__ __forceinline__ v16bf load_b_frag(const bf16_t* __restrict__ W,
                                             int ldw, int n0, int k0, int lane) {
  int n  = n0 + (lane & 15);
  int kb = k0 + ((lane >> 4) << 4);  // 0 or 16
  return *(const v16bf*)(W + (size_t)n * ldw + kb);
}

// --------------------------- utility kernels -------------------------------

__global__ void cast_f32_to_bf16(const float* __restrict__ src,
                                 bf16_t* __restrict__ dst, int n) {
  int i = blockIdx.x * blockDim.x + threadIdx.x;
  int stride = gridDim.x * blockDim.x;
  for (; i < n; i += stride) dst[i] = (bf16_t)src[i];
}

__global__ void zero_u32(unsigned* __restrict__ p, int n) {
  int i = blockIdx.x * blockDim.x + threadIdx.x;
  int stride = gridDim.x * blockDim.x;
  for (; i < n; i += stride) p[i] = 0u;
}

// --------------------------- input-to-hidden GEMM --------------------------
// out[m, n] = sum_k X[m,k] * W[n,k] + bias[n];  M=65536, N=2048, K=512.
// Block: 256 threads (8 waves), tile 128M x 64N; wave = 32x32 (2x2 WMMA tiles).

__global__ __launch_bounds__(256, 2)
void i2h_gemm(const bf16_t* __restrict__ X, const bf16_t* __restrict__ W,
              const float* __restrict__ bias, float* __restrict__ out) {
  const int lane = threadIdx.x & 31;
  const int wave = threadIdx.x >> 5;
  const int wm = wave & 3, wn = wave >> 2;
  const int m0 = blockIdx.y * 128 + wm * 32;
  const int n0 = blockIdx.x * 64  + wn * 32;

  v8f acc00 = {}, acc01 = {}, acc10 = {}, acc11 = {};
#pragma unroll 4
  for (int kc = 0; kc < D_; kc += 32) {
    v16bf a0 = load_a_frag(X, D_, m0,      kc, lane);
    v16bf a1 = load_a_frag(X, D_, m0 + 16, kc, lane);
    v16bf b0 = load_b_frag(W, D_, n0,      kc, lane);
    v16bf b1 = load_b_frag(W, D_, n0 + 16, kc, lane);
    acc00 = wmma_bf16(a0, b0, acc00);
    acc01 = wmma_bf16(a0, b1, acc01);
    acc10 = wmma_bf16(a1, b0, acc10);
    acc11 = wmma_bf16(a1, b1, acc11);
  }

  v8f accs[2][2] = {{acc00, acc01}, {acc10, acc11}};
#pragma unroll
  for (int i = 0; i < 2; ++i)
#pragma unroll
    for (int j = 0; j < 2; ++j) {
      int n  = n0 + j * 16 + (lane & 15);
      float bv = bias[n];
      int mb = m0 + i * 16 + ((lane >> 4) << 3);
      v8f a = accs[i][j];
#pragma unroll
      for (int r = 0; r < 8; ++r)
        out[(size_t)(mb + r) * G4_ + n] = a[r] + bv;
    }
}

// --------------------------- recurrent scan --------------------------------
// 16 blocks x 128 threads (4 waves). block = (batch-half bg, column-slice sl).
// Each wave owns 16 hidden columns; computes all 4 gate tiles for them.
// Wh slice [4][64][512] bf16 = 256 KB resident in LDS for the whole scan.

__device__ __forceinline__ float sigmoid_(float x) {
  return 1.0f / (1.0f + __expf(-x));
}

__global__ __launch_bounds__(128, 1)
void lstm_scan(const float* __restrict__ i2h,   // [B*T, 4H] f32 (bi folded in)
               const bf16_t* __restrict__ Wh,   // [4H, H] bf16
               const float* __restrict__ bh,    // [4H] f32
               bf16_t* __restrict__ hbuf,       // [2][32][512] bf16 (zeroed)
               bf16_t* __restrict__ y_bf,       // layer0: [B,T,H] bf16 (or null)
               float*  __restrict__ y_f32,      // layer1: [B,T,H] f32  (or null)
               float*  __restrict__ hT,         // [B,H]
               float*  __restrict__ cT,         // [B,H]
               unsigned* __restrict__ barrier_cnt) {  // [T] zeroed
  extern __shared__ bf16_t sWh[];  // [4 gates][64 cols][512 k]
  const int tid   = threadIdx.x;
  const int lane  = tid & 31;
  const int wave  = tid >> 5;          // 0..3
  const int bg    = blockIdx.x & 1;    // batch half (rows bg*16 .. +16)
  const int sl    = blockIdx.x >> 1;   // column slice 0..7
  const int jbase = sl * 64;
  const unsigned nblocks = gridDim.x;

  // Stage this WG's Wh slice into LDS once: 256 rows x 512 bf16 (256 KB).
  for (int i = tid; i < 256 * 64; i += 128) {
    int r = i >> 6;            // 0..255 = gate*64 + c
    int o = (i & 63) << 3;     // element offset, 8 bf16 = 16B chunks
    int gate = r >> 6, c = r & 63;
    *(v8bf*)(&sWh[((size_t)r << 9) + o]) =
        *(const v8bf*)(Wh + (((size_t)(gate * 512 + jbase + c)) << 9) + o);
  }
  __syncthreads();

  float creg[8];
#pragma unroll
  for (int r = 0; r < 8; ++r) creg[r] = 0.0f;

  const int jcol = jbase + wave * 16 + (lane & 15);  // this lane's hidden col
  const int mloc = (lane >> 4) << 3;                 // 0 or 8
  const int bcol = wave * 16 + (lane & 15);          // local LDS col for B frag

  for (int t = 0; t < T_; ++t) {
    const bf16_t* hprev = hbuf + (size_t)(t & 1) * (32 * 512)
                               + (size_t)bg * (16 * 512);
    bf16_t* hnext = hbuf + (size_t)((t + 1) & 1) * (32 * 512)
                         + (size_t)bg * (16 * 512);

    // Prefetch next timestep's i2h row region into cache.
    if (t + 1 < T_) {
      size_t nrow = ((size_t)(bg * 16 + mloc) * T_ + (t + 1)) * (size_t)G4_;
      __builtin_prefetch(&i2h[nrow + jcol], 0, 1);
    }

    v8f zero = {};
    v8f acc[4] = {zero, zero, zero, zero};
#pragma unroll 4
    for (int kc = 0; kc < H_; kc += 32) {
      v16bf a = load_a_frag(hprev, H_, 0, kc, lane);  // h_prev [16 x 512]
      int kb = kc + ((lane >> 4) << 4);
#pragma unroll
      for (int g = 0; g < 4; ++g) {
        v16bf b = *(const v16bf*)(&sWh[(((size_t)(g * 64 + bcol)) << 9) + kb]);
        acc[g] = wmma_bf16(a, b, acc[g]);
      }
    }

#pragma unroll
    for (int r = 0; r < 8; ++r) {
      int m = mloc + r;            // 0..15 local batch row
      int b = bg * 16 + m;
      size_t row = ((size_t)b * T_ + t) * (size_t)G4_;
      float gi = acc[0][r] + i2h[row + jcol]        + bh[jcol];
      float gf = acc[1][r] + i2h[row + 512 + jcol]  + bh[512 + jcol];
      float gc = acc[2][r] + i2h[row + 1024 + jcol] + bh[1024 + jcol];
      float go = acc[3][r] + i2h[row + 1536 + jcol] + bh[1536 + jcol];
      float ig = sigmoid_(gi);
      float fg = sigmoid_(gf);
      float cg = tanhf(gc);
      float og = sigmoid_(go);
      float cn = fg * creg[r] + ig * cg;
      float hn = og * tanhf(cn);
      creg[r] = cn;
      hnext[(size_t)m * H_ + jcol] = (bf16_t)hn;
      if (y_bf)  y_bf[((size_t)b * T_ + t) * H_ + jcol] = (bf16_t)hn;
      if (y_f32) y_f32[((size_t)b * T_ + t) * H_ + jcol] = hn;
      if (t == T_ - 1) {
        hT[(size_t)b * H_ + jcol] = hn;
        cT[(size_t)b * H_ + jcol] = cn;
      }
    }

    // Grid-wide barrier across the 16 co-resident blocks.
    __syncthreads();
    if (tid == 0) {
      __threadfence();
      atomicAdd(&barrier_cnt[t], 1u);
      volatile unsigned* vc = (volatile unsigned*)&barrier_cnt[t];
      while (*vc < nblocks) { }
      __threadfence();
    }
    __syncthreads();
  }
}

// --------------------------- host launch -----------------------------------

extern "C" void kernel_launch(void* const* d_in, const int* in_sizes, int n_in,
                              void* d_out, int out_size, void* d_ws, size_t ws_size,
                              hipStream_t stream) {
  (void)in_sizes; (void)n_in; (void)out_size; (void)ws_size;
  const float* x   = (const float*)d_in[0];
  const float* Wi0 = (const float*)d_in[1];
  const float* bi0 = (const float*)d_in[2];
  const float* Wh0 = (const float*)d_in[3];
  const float* bh0 = (const float*)d_in[4];
  const float* Wi1 = (const float*)d_in[5];
  const float* bi1 = (const float*)d_in[6];
  const float* Wh1 = (const float*)d_in[7];
  const float* bh1 = (const float*)d_in[8];
  float* out = (float*)d_out;

  // Workspace carving (256B aligned).
  char* ws = (char*)d_ws;
  size_t off = 0;
  auto take = [&](size_t bytes) -> char* {
    char* p = ws + off;
    off += (bytes + 255) & ~(size_t)255;
    return p;
  };
  bf16_t*  xb    = (bf16_t*)take((size_t)B_ * T_ * D_ * 2);
  bf16_t*  y0b   = (bf16_t*)take((size_t)B_ * T_ * H_ * 2);
  bf16_t*  Wi0b  = (bf16_t*)take((size_t)G4_ * D_ * 2);
  bf16_t*  Wh0b  = (bf16_t*)take((size_t)G4_ * H_ * 2);
  bf16_t*  Wi1b  = (bf16_t*)take((size_t)G4_ * H_ * 2);
  bf16_t*  Wh1b  = (bf16_t*)take((size_t)G4_ * H_ * 2);
  float*   i2h   = (float*)take((size_t)B_ * T_ * G4_ * 4);
  bf16_t*  hbuf  = (bf16_t*)take((size_t)2 * 32 * 512 * 2);
  unsigned* cnt  = (unsigned*)take((size_t)T_ * 4);

  const int ldsBytes = 4 * 64 * 512 * 2;  // 256 KB Wh slice
  hipFuncSetAttribute(reinterpret_cast<const void*>(lstm_scan),
                      hipFuncAttributeMaxDynamicSharedMemorySize, ldsBytes);

  // bf16 casts
  cast_f32_to_bf16<<<2048, 256, 0, stream>>>(x,   xb,   B_ * T_ * D_);
  cast_f32_to_bf16<<<256,  256, 0, stream>>>(Wi0, Wi0b, G4_ * D_);
  cast_f32_to_bf16<<<256,  256, 0, stream>>>(Wh0, Wh0b, G4_ * H_);
  cast_f32_to_bf16<<<256,  256, 0, stream>>>(Wi1, Wi1b, G4_ * H_);
  cast_f32_to_bf16<<<256,  256, 0, stream>>>(Wh1, Wh1b, G4_ * H_);

  // d_out layout: y1[B,T,H] | h[2,B,H] | c[2,B,H]
  float* hT0 = out + (size_t)B_ * T_ * H_;
  float* hT1 = hT0 + (size_t)B_ * H_;
  float* cT0 = hT1 + (size_t)B_ * H_;
  float* cT1 = cT0 + (size_t)B_ * H_;

  // ---- Layer 0 ----
  i2h_gemm<<<dim3(G4_ / 64, (B_ * T_) / 128), 256, 0, stream>>>(xb, Wi0b, bi0, i2h);
  zero_u32<<<64, 256, 0, stream>>>((unsigned*)hbuf, (2 * 32 * 512 * 2) / 4);
  zero_u32<<<8,  256, 0, stream>>>(cnt, T_);
  lstm_scan<<<16, 128, ldsBytes, stream>>>(i2h, Wh0b, bh0, hbuf,
                                           y0b, nullptr, hT0, cT0, cnt);

  // ---- Layer 1 ----
  i2h_gemm<<<dim3(G4_ / 64, (B_ * T_) / 128), 256, 0, stream>>>(y0b, Wi1b, bi1, i2h);
  zero_u32<<<64, 256, 0, stream>>>((unsigned*)hbuf, (2 * 32 * 512 * 2) / 4);
  zero_u32<<<8,  256, 0, stream>>>(cnt, T_);
  lstm_scan<<<16, 128, ldsBytes, stream>>>(i2h, Wh1b, bh1, hbuf,
                                           nullptr, out, hT1, cT1, cnt);
}
